// DisOrFuncf_34067680591904
// MI455X (gfx1250) — matrix operand order repaired and checked by reading, not compiled, not run on hardware
//
#include <hip/hip_runtime.h>
#include <math.h>
#include <stdint.h>

// Problem constants (match reference)
#define BATCH 256
#define NC    4
#define D_IN  784          // 49 * 16
#define D_H1  512          // 16 k-tiles of 32
#define D_H2  256          // 16 n-tiles of 16
#define NEG   0.2f
#define MROWS (BATCH * D_IN)       // 200704
#define MTILES (MROWS / 16)        // 12544
#define NTASKS (NC * MTILES)       // 50176 waves total

typedef __attribute__((ext_vector_type(16))) _Float16 v16h;
typedef __attribute__((ext_vector_type(8)))  float    v8f;

__device__ __forceinline__ float lrelu(float v) { return v > 0.f ? v : NEG * v; }

// ---- workspace layout (bytes) ----
// d1    : 256*512  f32  =  524288
// W1T   : 784*512  f32  = 1605632   (W1 transposed: [p][k], k-contiguous)
// Bpack : 512*256  f16  =  262144   (W2 in WMMA B-fragment order)
// fnew  : 4*256*784 f32 = 3211264
// fout  : 256 f32
#define OFF_D1    0u
#define OFF_W1T   524288u
#define OFF_BPACK 2129920u
#define OFF_FNEW  2392064u
#define OFF_FOUT  5603328u
// total ~5.6 MB of d_ws

// ------------------------------------------------------------------
// d1[b,o] = sample_x[b,:] . W1[o,:] + b1[o]     (256 x 512)
// ------------------------------------------------------------------
__global__ void d1_kernel(const float* __restrict__ x, const float* __restrict__ W1,
                          const float* __restrict__ b1, float* __restrict__ d1) {
  int i = blockIdx.x * blockDim.x + threadIdx.x;   // 131072 threads
  int b = i >> 9, o = i & 511;
  const float* xb = x + b * (NC * D_IN);           // x[b,0,:]
  const float* w  = W1 + o * D_IN;
  float acc = 0.f;
  for (int k = 0; k < D_IN; k += 4) {
    float4 xv = *(const float4*)(xb + k);
    float4 wv = *(const float4*)(w + k);
    acc += xv.x * wv.x + xv.y * wv.y + xv.z * wv.z + xv.w * wv.w;
  }
  d1[i] = acc + b1[o];
}

// W1T[p*512 + k] = W1[k*784 + p]  (k-contiguous rows for float4 A-side loads)
__global__ void packw1t_kernel(const float* __restrict__ W1, float* __restrict__ W1T) {
  int i = blockIdx.x * blockDim.x + threadIdx.x;   // 784*512 = 401408
  int p = i >> 9, k = i & 511;
  W1T[i] = W1[k * D_IN + p];
}

// Pack W2 (o-major, [n][k]) into WMMA 16x16x32 f16 B-fragment order:
// fragment(kt,nt): lane L holds column n = nt*16 + (L&15); element e (of 16
// packed halves = VGPR e/2, half e%2) holds k = kt*32 + (e>=8 ? 16 : 0) + (L>=16 ? 8 : 0) + (e&7).
// Flat index = ((kt*16 + nt)*32 + lane)*16 + e  -> each lane's 16 halves are
// 32 contiguous bytes (one v16h load).
__global__ void packb_kernel(const float* __restrict__ W2, _Float16* __restrict__ Bp) {
  int t = blockIdx.x * blockDim.x + threadIdx.x;   // 512*256 = 131072
  int e    = t & 15;
  int lane = (t >> 4) & 31;
  int nt   = (t >> 9) & 15;
  int kt   = t >> 13;
  int n = nt * 16 + (lane & 15);
  int k = kt * 32 + ((e >> 3) << 4) + ((lane >> 4) << 3) + (e & 7);
  Bp[t] = (_Float16)W2[n * D_H1 + k];              // B[k,n] = W2[n,k]
}

// ------------------------------------------------------------------
// Scalar path: fout[b] = sigmoid( W3 . lrelu(W2 . lrelu(d1[b]) + b2) + b3 )
// ------------------------------------------------------------------
__global__ void fout_kernel(const float* __restrict__ d1, const float* __restrict__ W2,
                            const float* __restrict__ b2, const float* __restrict__ W3,
                            const float* __restrict__ b3, float* __restrict__ fout) {
  __shared__ float red[256];
  int b = blockIdx.x, o = threadIdx.x;
  const float* db = d1 + b * D_H1;
  const float* w  = W2 + o * D_H1;
  float acc = 0.f;
  for (int k = 0; k < D_H1; ++k) acc += lrelu(db[k]) * w[k];
  float t = lrelu(acc + b2[o]);
  red[o] = t * W3[o];
  __syncthreads();
  for (int s = 128; s > 0; s >>= 1) { if (o < s) red[o] += red[o + s]; __syncthreads(); }
  if (o == 0) { float s = red[0] + b3[0]; fout[b] = 1.f / (1.f + expf(-s)); }
}

// Issue one 16KB B-slice (64 B per thread, 4 async b128 transfers) into LDS.
// ASYNCcnt tracks completion; async loads complete in order (ISA ch.10.4).
__device__ __forceinline__ void issue_b_slice(uint32_t lds_addr, uint64_t gbase,
                                              uint32_t voff) {
  asm volatile(
      "global_load_async_to_lds_b128 %0, %1, %2\n\t"
      "global_load_async_to_lds_b128 %0, %1, %2 offset:16\n\t"
      "global_load_async_to_lds_b128 %0, %1, %2 offset:32\n\t"
      "global_load_async_to_lds_b128 %0, %1, %2 offset:48"
      :: "v"(lds_addr), "v"(voff), "s"(gbase) : "memory");
}

// ------------------------------------------------------------------
// Main fused kernel: one wave = one level x one M-tile of 16 (b,p) rows.
// A rows generated on the fly: h = lrelu(d1[b,k] + a[b,p]*W1T[p,k]) -> f16;
// B fragments double-buffered in LDS via async global->LDS DMA (all 8 waves
// of the block share the same W2 fragments -> 8x less L2 traffic);
// 16x16x32 f16 WMMA, f32 accumulation, all 16 N-tiles in registers.
// ------------------------------------------------------------------
__global__ void __launch_bounds__(256, 1) mlp_levels_kernel(
    const float* __restrict__ x,   const float* __restrict__ d1g,
    const float* __restrict__ W1T, const _Float16* __restrict__ Bp,
    const float* __restrict__ b2,  const float* __restrict__ W3,
    const float* __restrict__ b3,  float* __restrict__ fnew) {
  __shared__ __align__(32) _Float16 Bs[2][8192];   // 2 x 16KB k-slices

  const int tid  = threadIdx.x;
  const int lane = tid & 31;
  const int wave = tid >> 5;
  const int task = blockIdx.x * 8 + wave;          // < 50176 exactly
  const int lvl   = task / MTILES;
  const int mtile = task - lvl * MTILES;

  // A-matrix row owned by this lane (ISA: lane&15 == M row)
  const int row0 = mtile * 16;
  const int m = lane & 15;
  const int r = row0 + m;
  const int b = r / D_IN;
  const int p = r - b * D_IN;
  const float a = (float)lvl - x[b * (NC * D_IN) + p];   // lvl - sample_x[b,p]
  // K-offset group for this lane half (A layout: lanes>=16 carry K+8 / K+24)
  const int kg = (lane >> 4) * 8;

  const float* db = d1g + b * D_H1 + kg;
  const float* wp = W1T + p * D_H1 + kg;

  const uint64_t gB   = (uint64_t)(uintptr_t)Bp;
  const uint32_t lds0 = (uint32_t)(uintptr_t)(&Bs[0][0]);
  const uint32_t lds1 = (uint32_t)(uintptr_t)(&Bs[1][0]);
  const uint32_t toff = (uint32_t)tid * 64u;

  // Prefetch k-slice 0 into buffer 0
  issue_b_slice(lds0 + toff, gB, toff);

  v8f acc[16];
  v8f zero = {0.f, 0.f, 0.f, 0.f, 0.f, 0.f, 0.f, 0.f};
#pragma unroll
  for (int nt = 0; nt < 16; ++nt) acc[nt] = zero;

  for (int kt = 0; kt < 16; ++kt) {
    const int kb = kt * 32;

    // Issue next slice into the other buffer (its previous readers finished
    // at the end-of-iteration barrier of kt-1).
    if (kt < 15) {
      uint32_t lbase = (((kt + 1) & 1) ? lds1 : lds0) + toff;
      issue_b_slice(lbase, gB, (uint32_t)(kt + 1) * 16384u + toff);
    }

    // Build A fragment while the async copy is in flight:
    // elements 0..7 -> K = kb+kg+0..7, 8..15 -> K = kb+16+kg+0..7
    float dd[16], ww[16];
    ((float4*)dd)[0] = *(const float4*)(db + kb);
    ((float4*)dd)[1] = *(const float4*)(db + kb + 4);
    ((float4*)dd)[2] = *(const float4*)(db + kb + 16);
    ((float4*)dd)[3] = *(const float4*)(db + kb + 20);
    ((float4*)ww)[0] = *(const float4*)(wp + kb);
    ((float4*)ww)[1] = *(const float4*)(wp + kb + 4);
    ((float4*)ww)[2] = *(const float4*)(wp + kb + 16);
    ((float4*)ww)[3] = *(const float4*)(wp + kb + 20);
    v16h af;
#pragma unroll
    for (int e = 0; e < 16; ++e) {
      float h = dd[e] + a * ww[e];
      h = h > 0.f ? h : NEG * h;                   // layer-1 leaky relu
      af[e] = (_Float16)h;
    }

    // Wait for this iteration's slice (async loads complete in order: with 8
    // outstanding, <=4 remaining means the 4 older ones have landed).
    if (kt < 15) asm volatile("s_wait_asynccnt 0x4" ::: "memory");
    else         asm volatile("s_wait_asynccnt 0x0" ::: "memory");
    __syncthreads();                               // slice visible to all waves

    const _Float16* bsl = &Bs[kt & 1][lane * 16];
#pragma unroll
    for (int nt = 0; nt < 16; ++nt) {
      v16h bf = *(const v16h*)(bsl + nt * 512);    // ds_load from shared slice
      acc[nt] = __builtin_amdgcn_wmma_f32_16x16x32_f16(
          /*neg_a=*/false, af, /*neg_b=*/false, bf,
          /*c_mod=*/(short)0, acc[nt], /*reuse_a=*/false, /*reuse_b=*/false);
    }
    __syncthreads();                               // done reading this buffer
  }

  // Epilogue. C layout: lane holds column n = lane&15; VGPR j holds row
  // M = (lane>=16 ? 8 : 0) + j == kg + j.
  float sj[8];
#pragma unroll
  for (int j = 0; j < 8; ++j) sj[j] = 0.f;
#pragma unroll
  for (int nt = 0; nt < 16; ++nt) {
    int nc = nt * 16 + m;
    float bb = b2[nc], w3 = W3[nc];
#pragma unroll
    for (int j = 0; j < 8; ++j) {
      float v = acc[nt][j] + bb;
      v = v > 0.f ? v : NEG * v;                   // layer-2 leaky relu
      sj[j] += v * w3;
    }
  }
  // Reduce across the 16 lanes that share each M-row group (wave32 shuffles)
#pragma unroll
  for (int j = 0; j < 8; ++j) {
    float v = sj[j];
    v += __shfl_xor(v, 1, 32);
    v += __shfl_xor(v, 2, 32);
    v += __shfl_xor(v, 4, 32);
    v += __shfl_xor(v, 8, 32);
    sj[j] = v;
  }
  if (m == 0) {
    const float b3v = b3[0];
#pragma unroll
    for (int j = 0; j < 8; ++j) {
      int rr = row0 + kg + j;
      int bb = rr / D_IN;
      int pp = rr - bb * D_IN;
      float s = sj[j] + b3v;
      fnew[(lvl * BATCH + bb) * D_IN + pp] = 1.f / (1.f + expf(-s));
    }
  }
}

// ------------------------------------------------------------------
// out[b] = inner + (fout - inner), inner = sum_{c,p} x[b,c,p]*(fnew[c]-fnew[3])
// ------------------------------------------------------------------
__global__ void final_kernel(const float* __restrict__ x, const float* __restrict__ fnew,
                             const float* __restrict__ fout, float* __restrict__ out) {
  __shared__ float red[256];
  int b = blockIdx.x, t = threadIdx.x;
  float acc = 0.f;
  for (int j = t; j < NC * D_IN; j += 256) {
    int c = j / D_IN, p = j - c * D_IN;
    float g = fnew[(c * BATCH + b) * D_IN + p] - fnew[(3 * BATCH + b) * D_IN + p];
    acc += x[b * NC * D_IN + j] * g;
  }
  red[t] = acc;
  __syncthreads();
  for (int s = 128; s > 0; s >>= 1) { if (t < s) red[t] += red[t + s]; __syncthreads(); }
  if (t == 0) { float inner = red[0]; out[b] = inner + (fout[b] - inner); }
}

extern "C" void kernel_launch(void* const* d_in, const int* in_sizes, int n_in,
                              void* d_out, int out_size, void* d_ws, size_t ws_size,
                              hipStream_t stream) {
  // inputs: x, is_train_g(=1), W1, b1, W2, b2, W3, b3
  const float* x  = (const float*)d_in[0];
  const float* W1 = (const float*)d_in[2];
  const float* b1 = (const float*)d_in[3];
  const float* W2 = (const float*)d_in[4];
  const float* b2 = (const float*)d_in[5];
  const float* W3 = (const float*)d_in[6];
  const float* b3 = (const float*)d_in[7];

  char* ws = (char*)d_ws;
  float*     d1   = (float*)(ws + OFF_D1);
  float*     W1T  = (float*)(ws + OFF_W1T);
  _Float16*  Bp   = (_Float16*)(ws + OFF_BPACK);
  float*     fnew = (float*)(ws + OFF_FNEW);
  float*     fout = (float*)(ws + OFF_FOUT);
  float*     out  = (float*)d_out;

  d1_kernel      <<<512,  256, 0, stream>>>(x, W1, b1, d1);
  packw1t_kernel <<<1568, 256, 0, stream>>>(W1, W1T);
  packb_kernel   <<<512,  256, 0, stream>>>(W2, Bp);
  fout_kernel    <<<256,  256, 0, stream>>>(d1, W2, b2, W3, b3, fout);
  // 50176 wave-tasks / 8 waves per 256-thread block = 6272 blocks
  mlp_levels_kernel<<<6272, 256, 0, stream>>>(x, d1, W1T, Bp, b2, W3, b3, fnew);
  final_kernel   <<<256,  256, 0, stream>>>(x, fnew, fout, out);
}
